// ECGLSTM_7052336300076
// MI455X (gfx1250) — matrix operand order repaired and checked by reading, not compiled, not run on hardware
//
#include <hip/hip_runtime.h>

typedef __attribute__((ext_vector_type(16))) _Float16 v16h;
typedef __attribute__((ext_vector_type(8)))  _Float16 v8h;
typedef __attribute__((ext_vector_type(8)))  float    v8f;

#define HSZ   64
#define BSZ   256
#define TSZ   5000
#define CHUNK 256

__device__ __forceinline__ float sigmoidf_(float x) {
    return __builtin_amdgcn_rcpf(1.0f + __expf(-x));
}
__device__ __forceinline__ float tanhf_(float x) {
    // tanh(x) = 1 - 2/(e^{2x}+1)
    return 1.0f - 2.0f * __builtin_amdgcn_rcpf(__expf(2.0f * x) + 1.0f);
}

// Load one 16x32 f16 A-fragment row for this lane from an LDS row of h.
// lane<16 -> M=lane, K = base..base+15 with base = 0; lane>=16 -> base = 16.
// `off` = kc*32 + 16*hib.
__device__ __forceinline__ v16h load_a_frag(const _Float16* row, int off) {
    v8h lo = *(const v8h*)(row + off);
    v8h hi = *(const v8h*)(row + off + 8);
    v16h a;
#pragma unroll
    for (int e = 0; e < 8; ++e) { a[e] = lo[e]; a[8 + e] = hi[e]; }
    return a;
}

extern "C" __global__ __launch_bounds__(128, 1)
void ecg_lstm_fused(const float* __restrict__ x,
                    const float* __restrict__ W_ih,
                    const float* __restrict__ W_hh,
                    const float* __restrict__ b_ih,
                    const float* __restrict__ b_hh,
                    const float* __restrict__ W_fc,
                    const float* __restrict__ b_fc,
                    float* __restrict__ out)
{
    __shared__ _Float16 hbuf[2][16][HSZ];   // double-buffered h_t (f16), 4 KB
    __shared__ float    xbuf[CHUNK][16];    // staged x chunk, transposed, 16 KB

    const int tid  = threadIdx.x;
    const int wave = tid >> 5;        // 0..3
    const int lane = tid & 31;
    const int ln   = lane & 15;       // N within tile / A-row M
    const int hib  = lane >> 4;       // half-wave select
    const int mrow = 8 * hib;         // C-fragment row base
    const int b0   = blockIdx.x * 16; // batch tile base

    // ---- preload recurrent weights as f16 B-fragments (stay in VGPRs) ----
    // wave w owns gate tiles {w, w+4, w+8, w+12}: all 4 gates for hidden
    // columns [16w, 16w+16).  B layout: lane<16 -> N=ln, K=0..15 of chunk;
    // lane>=16 -> N=ln, K=16..31 of chunk.  B[k,j] = W_hh[j,k] (row-contig).
    v16h wb[4][2];
    float wi[4], bs[4];
#pragma unroll
    for (int g = 0; g < 4; ++g) {
        const int j = (4 * g + wave) * 16 + ln;     // gate unit in [0,256)
        wi[g] = W_ih[j];
        bs[g] = b_ih[j] + b_hh[j];
#pragma unroll
        for (int kc = 0; kc < 2; ++kc) {
            const float* src = W_hh + j * HSZ + kc * 32 + hib * 16;
            v16h f;
#pragma unroll
            for (int e = 0; e < 16; ++e) f[e] = (_Float16)src[e];
            wb[g][kc] = f;
        }
    }

    // ---- zero initial state ----
    float creg[8];
#pragma unroll
    for (int r = 0; r < 8; ++r) creg[r] = 0.0f;
    for (int i = tid; i < 16 * HSZ; i += 128)
        hbuf[0][i >> 6][i & 63] = (_Float16)0.0f;
    __syncthreads();

    const int hc = wave * 16 + ln;    // hidden column owned by this lane

    // ---- sequential time loop ----
    for (int t0 = 0; t0 < TSZ; t0 += CHUNK) {
        // stage x[b0..b0+15, t0..t0+CHUNK) into LDS (transposed: xbuf[t][m])
        {
            const int m   = tid >> 3;
            const int tt0 = tid & 7;
#pragma unroll 4
            for (int k = 0; k < CHUNK / 8; ++k) {
                const int tt = tt0 + 8 * k;
                const int t  = t0 + tt;
                xbuf[tt][m] = (t < TSZ) ? x[(b0 + m) * TSZ + t] : 0.0f;
            }
        }
        __syncthreads();
        const int nstep = (TSZ - t0 < CHUNK) ? (TSZ - t0) : CHUNK;

        // software-pipelined accumulator init for step 0 of this chunk:
        // gates start at x*W_ih + (b_ih + b_hh)
        v8f acc[4];
        {
            const float* xr = &xbuf[0][mrow];
#pragma unroll
            for (int g = 0; g < 4; ++g)
#pragma unroll
                for (int r = 0; r < 8; ++r) acc[g][r] = xr[r] * wi[g] + bs[g];
        }

        for (int s = 0; s < nstep; ++s) {
            const int t   = t0 + s;
            const int cur = t & 1, nxt = cur ^ 1;

            // A fragments: current h (16x64 f16) from LDS
            const _Float16* hrow = &hbuf[cur][ln][0];
            const v16h a0 = load_a_frag(hrow, 2 * mrow);        // K 0..31
            const v16h a1 = load_a_frag(hrow, 2 * mrow + 32);   // K 32..63

            // gates += h @ W_hh^T   (4 gate tiles, K=64 in two WMMAs)
            v8f gate[4];
#pragma unroll
            for (int g = 0; g < 4; ++g) {
                v8f a = __builtin_amdgcn_wmma_f32_16x16x32_f16(
                        false, a0, false, wb[g][0], (short)0, acc[g], false, false);
                a = __builtin_amdgcn_wmma_f32_16x16x32_f16(
                        false, a1, false, wb[g][1], (short)0, a, false, false);
                gate[g] = a;
            }

            // LSTM cell update: this lane owns (rows mrow..mrow+7, col hc)
#pragma unroll
            for (int r = 0; r < 8; ++r) {
                const float iv = sigmoidf_(gate[0][r]);
                const float fv = sigmoidf_(gate[1][r]);
                const float gv = tanhf_(gate[2][r]);
                const float ov = sigmoidf_(gate[3][r]);
                const float c  = fv * creg[r] + iv * gv;
                creg[r] = c;
                hbuf[nxt][mrow + r][hc] = (_Float16)(ov * tanhf_(c));
            }

            // ---- CDNA5 split barrier: drain LDS ops (A-reads + h-stores),
            // arrive, then overlap next step's independent accumulator init
            // with other waves' arrival before waiting. ----
            asm volatile("s_wait_dscnt 0x0\n\t"
                         "s_barrier_signal -1" ::: "memory");
            if (s + 1 < nstep) {
                const float* xr = &xbuf[s + 1][mrow];
#pragma unroll
                for (int g = 0; g < 4; ++g)
#pragma unroll
                    for (int r = 0; r < 8; ++r)
                        acc[g][r] = xr[r] * wi[g] + bs[g];
            }
            asm volatile("s_barrier_wait -1" ::: "memory");
        }
    }

    // ---- final FC: out[16x128] = relu(h_last @ W_fc^T + b_fc) ----
    const int fb = TSZ & 1;                  // buffer holding h_T
    const _Float16* hrow = &hbuf[fb][ln][0];
    const v16h a0 = load_a_frag(hrow, 2 * mrow);
    const v16h a1 = load_a_frag(hrow, 2 * mrow + 32);
#pragma unroll
    for (int g = 0; g < 2; ++g) {
        const int j = (4 * g + wave) * 16 + ln;   // output column in [0,128)
        v16h fw[2];
#pragma unroll
        for (int kc = 0; kc < 2; ++kc) {
            const float* src = W_fc + j * HSZ + kc * 32 + hib * 16;
            v16h f;
#pragma unroll
            for (int e = 0; e < 16; ++e) f[e] = (_Float16)src[e];
            fw[kc] = f;
        }
        const float bj = b_fc[j];
        v8f a;
#pragma unroll
        for (int r = 0; r < 8; ++r) a[r] = bj;
        a = __builtin_amdgcn_wmma_f32_16x16x32_f16(
                false, a0, false, fw[0], (short)0, a, false, false);
        a = __builtin_amdgcn_wmma_f32_16x16x32_f16(
                false, a1, false, fw[1], (short)0, a, false, false);
#pragma unroll
        for (int r = 0; r < 8; ++r)
            out[(b0 + mrow + r) * 128 + j] = fmaxf(a[r], 0.0f);
    }
}

extern "C" void kernel_launch(void* const* d_in, const int* in_sizes, int n_in,
                              void* d_out, int out_size, void* d_ws, size_t ws_size,
                              hipStream_t stream) {
    const float* x    = (const float*)d_in[0];
    const float* W_ih = (const float*)d_in[1];
    const float* W_hh = (const float*)d_in[2];
    const float* b_ih = (const float*)d_in[3];
    const float* b_hh = (const float*)d_in[4];
    const float* W_fc = (const float*)d_in[5];
    const float* b_fc = (const float*)d_in[6];
    float* out = (float*)d_out;

    // 16 workgroups (one 16-row batch tile each), 4 waves (128 threads) per WG
    ecg_lstm_fused<<<BSZ / 16, 128, 0, stream>>>(x, W_ih, W_hh, b_ih, b_hh,
                                                 W_fc, b_fc, out);
}
